// MambaEncoder_28381143892543
// MI455X (gfx1250) — matrix operand order repaired and checked by reading, not compiled, not run on hardware
//
#include <hip/hip_runtime.h>
#include <hip/hip_bf16.h>
#include <cstdint>

// ---------------------------------------------------------------------------
// MI455X (gfx1250) Mamba-encoder pipeline.
// Bandwidth-bound (~5 GB streamed @ 23.3 TB/s); GEMMs use bf16 WMMA
// (v_wmma_f32_16x16x32_bf16). Weight tiles are staged into LDS with the
// CDNA5 async engine (global_load_async_to_lds_b128 / ASYNCcnt); B-fragments
// are double-buffered so ds_load latency hides behind WMMA issue.
// ---------------------------------------------------------------------------

typedef __attribute__((ext_vector_type(16))) __bf16       v16bf;
typedef __attribute__((ext_vector_type(8)))  float        v8f;
typedef __attribute__((ext_vector_type(4)))  unsigned int v4u;

static constexpr int NN = 512;
static constexpr int KK = 512;
static constexpr int FF = 96;
static constexpr int DM = 256;
static constexpr int DS = 256;
static constexpr int LL = 2;
static constexpr float EPSC = 1e-5f;
static constexpr int TT = NN * KK; // 262144 tokens

__device__ __forceinline__ unsigned short f2bf(float f) {
  union { float f; unsigned u; } v; v.f = f;
  unsigned r = v.u + 0x7FFFu + ((v.u >> 16) & 1u); // RNE
  return (unsigned short)(r >> 16);
}

// Low 32 bits of a generic pointer to __shared__ = LDS byte offset.
__device__ __forceinline__ unsigned lds_off32(const void* p) {
  return (unsigned)(uintptr_t)p;
}

union FragB { v4u q[2]; v16bf v; };

enum { EPI_PLAIN = 0, EPI_SILU = 1, EPI_RES = 2, EPI_DXG = 3 };

// ---------------------------------------------------------------------------
// Tiled GEMM: O[t,e] = epi( sum_k A[t,k] * Wbf[e,k] + bias[e] ), E == 256.
// Block: 256 thr (8 waves), tile 128 rows x 256 cols, K-step 32.
// Wave w owns rows [w*16, w*16+16) and all 16 n-tiles (16 v8f accumulators).
// LDS rows padded to 40 halfwords (80B) -> conflict-free b128 frag loads.
// ---------------------------------------------------------------------------
template <int MODE>
__global__ __launch_bounds__(256) void gemm_bf16_kernel(
    const float* __restrict__ A, int Kdim,
    const unsigned short* __restrict__ Wbf,   // [256][Kdim] bf16
    const float* __restrict__ bias,           // [256] or null (DXG)
    const float* resid,                       // may alias O (same-index RMW)
    const float* __restrict__ dvec,           // [256] (DXG)
    float* O) {
  constexpr int SAS = 40;
  __shared__ unsigned short sA[128 * SAS];
  __shared__ unsigned short sW[256 * SAS];

  const int tid  = threadIdx.x;
  const int wid  = tid >> 5;
  const int lane = tid & 31;
  const int half = lane >> 4;
  const int lm   = lane & 15;
  const long rowBase = (long)blockIdx.x * 128;

  const v8f vzero = {0.f, 0.f, 0.f, 0.f, 0.f, 0.f, 0.f, 0.f};
  v8f acc[16];
#pragma unroll
  for (int i = 0; i < 16; ++i) acc[i] = vzero;

  const int nkt = Kdim >> 5;
  for (int kt = 0; kt < nkt; ++kt) {
    const int k0 = kt << 5;
    // --- stage W tile via async DMA: 256 rows x 32 bf16 (ASYNCcnt path) ---
#pragma unroll
    for (int i = 0; i < 4; ++i) {
      const int slot = tid + i * 256;
      const int r  = slot >> 2;
      const int c8 = (slot & 3) << 3;
      const unsigned short* gp = Wbf + (size_t)r * Kdim + k0 + c8;
      const unsigned dst = lds_off32(&sW[r * SAS + c8]);
      asm volatile("global_load_async_to_lds_b128 %0, %1, off"
                   :: "v"(dst), "v"((unsigned long long)(uintptr_t)gp)
                   : "memory");
    }
    // --- stage A tile: 128 rows x 32 f32 -> bf16 in LDS -------------------
#pragma unroll
    for (int i = 0; i < 4; ++i) {
      const int slot = tid + i * 256;
      const int r  = slot >> 3;
      const int c4 = (slot & 7) << 2;
      const float4 v = *(const float4*)(A + (rowBase + r) * Kdim + k0 + c4);
      const unsigned u0 = (unsigned)f2bf(v.x) | ((unsigned)f2bf(v.y) << 16);
      const unsigned u1 = (unsigned)f2bf(v.z) | ((unsigned)f2bf(v.w) << 16);
      *(uint2*)(&sA[r * SAS + c4]) = make_uint2(u0, u1);
      if (k0 + 32 < Kdim)  // prefetch next k-tile (global_prefetch_b8)
        __builtin_prefetch(A + (rowBase + r) * Kdim + k0 + 32 + c4, 0, 1);
    }
    asm volatile("s_wait_asynccnt 0x0" ::: "memory");
    __syncthreads();

    // A frag (16x32 bf16): lane lm row; half selects K {0..7,16..23}/{8..15,24..31}
    FragB af;
    {
      const int base = (wid * 16 + lm) * SAS + half * 8;
      af.q[0] = *(const v4u*)&sA[base];
      af.q[1] = *(const v4u*)&sA[base + 16];
    }
    // B frags double-buffered: loads for nt+1 issue before WMMA nt, so the
    // dscnt wait before each WMMA tolerates 2 in-flight ds_loads.
    FragB bfr[2];
    {
      const int base = lm * SAS + half * 8;
      bfr[0].q[0] = *(const v4u*)&sW[base];
      bfr[0].q[1] = *(const v4u*)&sW[base + 16];
    }
#pragma unroll
    for (int nt = 0; nt < 16; ++nt) {
      if (nt < 15) {
        const int base = ((nt + 1) * 16 + lm) * SAS + half * 8;
        bfr[(nt + 1) & 1].q[0] = *(const v4u*)&sW[base];
        bfr[(nt + 1) & 1].q[1] = *(const v4u*)&sW[base + 16];
      }
      acc[nt] = __builtin_amdgcn_wmma_f32_16x16x32_bf16(
          false, af.v, false, bfr[nt & 1].v, (short)0, acc[nt], false, false);
    }
    __syncthreads();
  }

  // --- epilogue: D layout lane lm = col, VGPR r -> row half*8+r ------------
#pragma unroll
  for (int nt = 0; nt < 16; ++nt) {
    const int n = nt * 16 + lm;
    float bv = 0.f, dv = 0.f;
    if (MODE != EPI_DXG) bv = bias[n];
    if (MODE == EPI_DXG) dv = dvec[n];
#pragma unroll
    for (int r = 0; r < 8; ++r) {
      const long row = rowBase + wid * 16 + half * 8 + r;
      const long idx = row * 256 + n;
      float v = acc[nt][r] + bv;
      if (MODE == EPI_SILU) v = v / (1.f + __expf(-v));
      if (MODE == EPI_RES)  v += resid[idx];
      if (MODE == EPI_DXG)  v += dv * resid[idx];
      O[idx] = v;
    }
  }
}

// --------------------------- LayerNorm (+GELU) -----------------------------
__global__ __launch_bounds__(256) void ln_kernel(
    const float* in, const float* __restrict__ g, const float* __restrict__ b,
    float* out, int do_gelu) {
  const int wid = threadIdx.x >> 5, lane = threadIdx.x & 31;
  const long t = (long)blockIdx.x * 8 + wid;
  const float* p = in + t * DM + lane * 8;
  float xv[8];
  *(float4*)&xv[0] = *(const float4*)p;
  *(float4*)&xv[4] = *(const float4*)(p + 4);
  float s = 0.f, s2 = 0.f;
#pragma unroll
  for (int i = 0; i < 8; ++i) { s += xv[i]; s2 += xv[i] * xv[i]; }
#pragma unroll
  for (int off = 16; off; off >>= 1) {
    s  += __shfl_xor(s,  off, 32);
    s2 += __shfl_xor(s2, off, 32);
  }
  const float mu = s * (1.f / DM);
  const float rs = rsqrtf(s2 * (1.f / DM) - mu * mu + EPSC);
  float ov[8];
#pragma unroll
  for (int i = 0; i < 8; ++i) {
    const int c = lane * 8 + i;
    float y = (xv[i] - mu) * rs * g[c] + b[c];
    if (do_gelu) y = 0.5f * y * (1.f + erff(y * 0.70710678118f));
    ov[i] = y;
  }
  float* q = out + t * DM + lane * 8;
  *(float4*)q       = *(const float4*)&ov[0];
  *(float4*)(q + 4) = *(const float4*)&ov[4];
}

// --------------------------- SSM scan (in place) ---------------------------
// state carried ACROSS layers (reference reuses `state` between layers).
__global__ __launch_bounds__(256) void scan_kernel(
    float* s, const float* __restrict__ decay, float* state, int first) {
  const int tid = blockIdx.x * 256 + threadIdx.x; // 0..131071 (n*256+c)
  const int n = tid >> 8, c = tid & 255;
  const float d = decay[c];
  float st = first ? 0.f : state[tid];
  long base = (long)n * KK * 256 + c;
  for (int k = 0; k < KK; ++k) {
    st = st * d + s[base];
    s[base] = st;
    base += 256;
  }
  state[tid] = st;
}

__global__ void decay_kernel(const float* __restrict__ A_log,
                             float* __restrict__ decay) {
  const int d = threadIdx.x; // 256 threads, 1 block
  float s = 0.f;
  for (int i = 0; i < DS; ++i) s += expf(A_log[d * DS + i]);
  decay[d] = expf(-s * (1.f / DS));
}

// ------------------------ weight fp32 -> bf16 prep -------------------------
__global__ void convw_kernel(const float* __restrict__ src,
                             unsigned short* __restrict__ dst, int n) {
  const int i = blockIdx.x * 256 + threadIdx.x;
  if (i < n) dst[i] = f2bf(src[i]);
}
__global__ void convwT_kernel(const float* __restrict__ src,
                              unsigned short* __restrict__ dst, int R, int C) {
  const int i = blockIdx.x * 256 + threadIdx.x;
  if (i < R * C) { const int r = i / C, c = i % C; dst[c * R + r] = f2bf(src[i]); }
}

// ---------------------- head: pool + GEMV + LN + GELU ----------------------
__global__ __launch_bounds__(256) void head_kernel(
    const float* __restrict__ h, const float* __restrict__ w,
    const float* __restrict__ b, const float* __restrict__ g,
    const float* __restrict__ bb, float* __restrict__ out) {
  __shared__ float pooled[256];
  __shared__ float red[256];
  const int n = blockIdx.x, e = threadIdx.x;
  pooled[e] = h[((long)n * KK + (KK - 1)) * 256 + e];
  __syncthreads();
  float acc = b[e];
#pragma unroll 4
  for (int d2 = 0; d2 < 256; ++d2) acc += pooled[d2] * w[e * 256 + d2];
  red[e] = acc; __syncthreads();
  for (int off = 128; off; off >>= 1) { if (e < off) red[e] += red[e + off]; __syncthreads(); }
  const float mu = red[0] * (1.f / 256.f); __syncthreads();
  red[e] = acc * acc; __syncthreads();
  for (int off = 128; off; off >>= 1) { if (e < off) red[e] += red[e + off]; __syncthreads(); }
  const float var = red[0] * (1.f / 256.f) - mu * mu;
  float y = (acc - mu) * rsqrtf(var + EPSC) * g[e] + bb[e];
  y = 0.5f * y * (1.f + erff(y * 0.70710678118f));
  out[(long)n * 256 + e] = y;
}

// ---------------------------------------------------------------------------
extern "C" void kernel_launch(void* const* d_in, const int* in_sizes, int n_in,
                              void* d_out, int out_size, void* d_ws, size_t ws_size,
                              hipStream_t stream) {
  const float* x        = (const float*)d_in[0];
  const float* in_w     = (const float*)d_in[1];
  const float* in_b     = (const float*)d_in[2];
  const float* in_ln_g  = (const float*)d_in[3];
  const float* in_ln_b  = (const float*)d_in[4];
  const float* ln_g     = (const float*)d_in[5];
  const float* ln_b     = (const float*)d_in[6];
  const float* gate_w   = (const float*)d_in[7];
  const float* gate_b   = (const float*)d_in[8];
  const float* sp_w     = (const float*)d_in[9];
  const float* sp_b     = (const float*)d_in[10];
  const float* op_w     = (const float*)d_in[11];
  const float* op_b     = (const float*)d_in[12];
  const float* A_log    = (const float*)d_in[13];
  const float* Dvec     = (const float*)d_in[14];
  const float* out_w    = (const float*)d_in[15];
  const float* out_b    = (const float*)d_in[16];
  const float* out_ln_g = (const float*)d_in[17];
  const float* out_ln_b = (const float*)d_in[18];
  float* out = (float*)d_out;

  const size_t TD = (size_t)TT * DM; // 67,108,864 floats per buffer
  float* h  = (float*)d_ws;
  float* z  = h + TD;   // also reused as y
  float* xg = z + TD;
  float* sb = xg + TD;  // s, overwritten by scan states
  unsigned short* wbf     = (unsigned short*)(sb + TD);
  unsigned short* inw_bf  = wbf;              // 256*96
  unsigned short* gate_bf = wbf + 32768;      // 256*256 each below
  unsigned short* sp_bf   = gate_bf + 65536;
  unsigned short* spT_bf  = sp_bf + 65536;
  unsigned short* op_bf   = spT_bf + 65536;
  float* decayb = (float*)(op_bf + 65536);    // 256
  float* stbuf  = decayb + 256;               // 512*256 carried SSM state

  const dim3 blk(256);
  const int gemmGrid = TT / 128; // 2048
  const int lnGrid   = TT / 8;   // 32768

  // in-projection: h = gelu(LN(x @ in_w^T + in_b))
  convw_kernel<<<(DM * FF + 255) / 256, blk, 0, stream>>>(in_w, inw_bf, DM * FF);
  gemm_bf16_kernel<EPI_PLAIN><<<gemmGrid, blk, 0, stream>>>(
      x, FF, inw_bf, in_b, nullptr, nullptr, h);
  ln_kernel<<<lnGrid, blk, 0, stream>>>(h, in_ln_g, in_ln_b, h, 1);

  for (int l = 0; l < LL; ++l) {
    const float* gw = gate_w + (size_t)l * DM * DM;
    const float* sw = sp_w + (size_t)l * DS * DM;
    const float* ow = op_w + (size_t)l * DM * DM;
    convw_kernel<<<256, blk, 0, stream>>>(gw, gate_bf, DM * DM);
    convw_kernel<<<256, blk, 0, stream>>>(sw, sp_bf, DS * DM);
    convwT_kernel<<<256, blk, 0, stream>>>(sw, spT_bf, DS, DM);
    convw_kernel<<<256, blk, 0, stream>>>(ow, op_bf, DM * DM);
    decay_kernel<<<1, blk, 0, stream>>>(A_log + (size_t)l * DM * DS, decayb);

    // z = LN(h); xg = silu(z @ gate_w^T + gate_b); s = xg @ sp_w^T + sp_b
    ln_kernel<<<lnGrid, blk, 0, stream>>>(h, ln_g + l * DM, ln_b + l * DM, z, 0);
    gemm_bf16_kernel<EPI_SILU><<<gemmGrid, blk, 0, stream>>>(
        z, DM, gate_bf, gate_b + l * DM, nullptr, nullptr, xg);
    gemm_bf16_kernel<EPI_PLAIN><<<gemmGrid, blk, 0, stream>>>(
        xg, DM, sp_bf, sp_b + l * DS, nullptr, nullptr, sb);
    // sequential scan along K (in place), state carried across layers
    scan_kernel<<<512, blk, 0, stream>>>(sb, decayb, stbuf, l == 0 ? 1 : 0);
    // y = states @ sp_w + D*xg ; h = y @ op_w^T + op_b + h
    gemm_bf16_kernel<EPI_DXG><<<gemmGrid, blk, 0, stream>>>(
        sb, DS, spT_bf, nullptr, xg, Dvec + l * DM, z);
    gemm_bf16_kernel<EPI_RES><<<gemmGrid, blk, 0, stream>>>(
        z, DM, op_bf, op_b + l * DM, h, nullptr, h);
  }

  head_kernel<<<NN, blk, 0, stream>>>(h, out_w, out_b, out_ln_g, out_ln_b, out);
  (void)in_sizes; (void)n_in; (void)out_size; (void)ws_size;
}
